// seq_selfattn_37134287241225
// MI455X (gfx1250) — matrix operand order repaired and checked by reading, not compiled, not run on hardware
//
#include <hip/hip_runtime.h>
#include <hip/hip_bf16.h>

// ---------------------------------------------------------------------------
// MI455X (gfx1250) fused "sequence self-attention" pipeline, round 4.
//
// Round 3 -> 4: fix the async-to-LDS builtin call. The gfx1250 builtin
// signature is (v4i AS1*, v4i AS3*, imm offset, imm cpol) — cast through a
// generic v4i* (reinterpret) then to the address-space pointer (addrspacecast).
// ---------------------------------------------------------------------------

typedef __attribute__((ext_vector_type(16))) _Float16 v16h;
typedef __attribute__((ext_vector_type(8)))  _Float16 v8h;
typedef __attribute__((ext_vector_type(8)))  float    v8f;
typedef __attribute__((ext_vector_type(4)))  int      v4i;

#define DIM   512
#define HEADS 8
#define CH    64          // DIM / HEADS
#define NSEQ  2048
#define BATCH 4
#define C3    (3 * DIM)   // 1536

// --------------------------- async-to-LDS support --------------------------
#if __has_builtin(__builtin_amdgcn_global_load_async_to_lds_b128)
#define HAVE_ASYNC_LDS 1
typedef __attribute__((address_space(1))) v4i gv4i_t;
typedef __attribute__((address_space(3))) v4i lv4i_t;
#else
#define HAVE_ASYNC_LDS 0
#endif

// Copy 16 bytes global -> LDS (async when the gfx1250 builtin is available).
__device__ __forceinline__ void async_copy_b128(_Float16* lds_dst,
                                                const _Float16* gsrc) {
#if HAVE_ASYNC_LDS
    __builtin_amdgcn_global_load_async_to_lds_b128(
        (gv4i_t*)(v4i*)gsrc, (lv4i_t*)(v4i*)lds_dst, 0, 0);
#else
    *(v8h*)lds_dst = *(const v8h*)gsrc;
#endif
}

// Wait for outstanding async-LDS copies, then workgroup barrier.
__device__ __forceinline__ void wait_async_barrier() {
#if HAVE_ASYNC_LDS
#if __has_builtin(__builtin_amdgcn_s_wait_asynccnt)
    __builtin_amdgcn_s_wait_asynccnt(0);
#else
    asm volatile("s_wait_asynccnt 0x0" ::: "memory");
#endif
#endif
    __syncthreads();
}

// ----------------------- WMMA fragment index helpers -----------------------
// Fragment layout used here (wave32):
//  A 16x32 f16 : row M = lane&15 ; elem e -> K = g*16+e (contiguous; code's choice)
//  B 32x16 f16 : col N = lane&15 ; elem e -> K = g*16+e (contiguous; code's choice)
//  C/D 16x16 f32: col N = lane&15 ; VGPR r -> row M = r + 8*g
__device__ __forceinline__ v16h load_a_frag(const _Float16* rowp, int g) {
    return *(const v16h*)(rowp + g * 16);
}
__device__ __forceinline__ v16h load_b_frag(const _Float16* colp, int g) {
    return *(const v16h*)(colp + g * 16);
}

// ---------------------------------------------------------------------------
// Elementwise f32 -> f16 convert (weights)
// ---------------------------------------------------------------------------
__global__ void cvt_f16_flat(const float* __restrict__ in,
                             _Float16* __restrict__ out, int count) {
    int i = blockIdx.x * blockDim.x + threadIdx.x;
    if (i < count) out[i] = (_Float16)in[i];
}

// ---------------------------------------------------------------------------
// x [b][c][n] f32  ->  Xt [b][n][c] f16   (output-coalesced)
// ---------------------------------------------------------------------------
__global__ void transpose_cvt_x(const float* __restrict__ x,
                                _Float16* __restrict__ Xt) {
    size_t idx = (size_t)blockIdx.x * blockDim.x + threadIdx.x;
    const size_t total = (size_t)BATCH * NSEQ * DIM;
    if (idx >= total) return;
    const int c = (int)(idx % DIM);
    size_t t    = idx / DIM;
    const int n = (int)(t % NSEQ);
    const int b = (int)(t / NSEQ);
    Xt[idx] = (_Float16)x[((size_t)b * DIM + c) * NSEQ + n];
}

// ---------------------------------------------------------------------------
// Pointwise GEMM: Y[b,o,n] = sum_c W16[o,c] * Xt[b,n,c] + bias[o]   (f32 out)
// 4 waves per block -> 64(o) x 128(n) tile. X tile (128 n x 32 k, 8KB f16)
// double-buffered in LDS via async copies and shared by all 4 waves.
// ---------------------------------------------------------------------------
__global__ void gemm_pw_wmma(const _Float16* __restrict__ W16,
                             const float* __restrict__ bias,
                             const _Float16* __restrict__ Xt,
                             float* __restrict__ Y,
                             int O, int C, int Ncols) {
    const int b     = blockIdx.z;
    const int tid   = threadIdx.x;        // 0..127
    const int wave  = tid >> 5;
    const int lane  = tid & 31;
    const int li    = lane & 15;
    const int g     = (lane >> 4) & 1;
    const int obase = blockIdx.x * 64 + wave * 16;
    const int nbase = blockIdx.y * 128;

    const _Float16* Xb   = Xt + (size_t)b * Ncols * C;          // [n][C]
    float*          Yb   = Y + (size_t)b * O * Ncols;           // [o][n]
    const _Float16* Wrow = W16 + (size_t)(obase + li) * C;

    __shared__ alignas(32) _Float16 xtile[2][128 * 32];         // [n][k] 8KB x2

    // Stage X[nbase..nbase+127][kk..kk+31] into xtile[buf].
    // 512 chunks of 16B; each of 128 threads copies 4 chunks.
    auto stage = [&](int buf, int kk) {
#pragma unroll
        for (int u = 0; u < 4; ++u) {
            const int chunk = tid * 4 + u;            // 0..511
            const int row   = chunk >> 2;             // n offset in tile
            const int off   = (chunk & 3) * 8;        // half offset in row
            async_copy_b128(&xtile[buf][row * 32 + off],
                            Xb + (size_t)(nbase + row) * C + kk + off);
        }
    };

    v8f acc[8];
#pragma unroll
    for (int t = 0; t < 8; ++t) acc[t] = v8f{};

    stage(0, 0);
    wait_async_barrier();

    for (int kk = 0; kk < C; kk += 32) {
        const int cur = (kk >> 5) & 1;
        if (kk + 32 < C) stage(cur ^ 1, kk + 32);     // prefetch next K tile

        const v16h a = load_a_frag(Wrow + kk, g);     // 2x global_load_b128
#pragma unroll
        for (int t = 0; t < 8; ++t) {
            const v16h bf = load_b_frag(&xtile[cur][(t * 16 + li) * 32], g);
            acc[t] = __builtin_amdgcn_wmma_f32_16x16x32_f16(
                false, a, false, bf, (short)0, acc[t], false, false);
        }
        wait_async_barrier();                         // next tile ready
    }

#pragma unroll
    for (int t = 0; t < 8; ++t) {
        const int n = nbase + t * 16 + li;
#pragma unroll
        for (int r = 0; r < 8; ++r) {
            const int o = obase + r + 8 * g;
            Yb[(size_t)o * Ncols + n] = acc[t][r] + bias[o];
        }
    }
}

// ---------------------------------------------------------------------------
// Depthwise conv k=3 pad=1 over n, + bias, split into q0/k0/v0 (f32)
// ---------------------------------------------------------------------------
__global__ void dwconv_split(const float* __restrict__ qkv,
                             const float* __restrict__ wdw,
                             const float* __restrict__ bdw,
                             float* __restrict__ q0,
                             float* __restrict__ k0,
                             float* __restrict__ v0) {
    size_t idx = (size_t)blockIdx.x * blockDim.x + threadIdx.x;
    const size_t total = (size_t)BATCH * C3 * NSEQ;
    if (idx >= total) return;
    const int n = (int)(idx % NSEQ);
    size_t t    = idx / NSEQ;
    const int c = (int)(t % C3);
    const int b = (int)(t / C3);

    const float* row = qkv + ((size_t)b * C3 + c) * NSEQ;
    const float w0 = wdw[c * 3 + 0];
    const float w1 = wdw[c * 3 + 1];
    const float w2 = wdw[c * 3 + 2];
    float acc = bdw[c] + row[n] * w1;
    if (n > 0)        acc += row[n - 1] * w0;
    if (n < NSEQ - 1) acc += row[n + 1] * w2;

    const int which = c / DIM;
    const int cc    = c % DIM;
    float* dst = (which == 0) ? q0 : (which == 1) ? k0 : v0;
    dst[((size_t)b * DIM + cc) * NSEQ + n] = acc;
}

// ---------------------------------------------------------------------------
// q row L2 norms over the sequence axis
// ---------------------------------------------------------------------------
__global__ void qnorm_kernel(const float* __restrict__ q0,
                             float* __restrict__ qnorm) {
    const int bc = blockIdx.x;              // b*DIM + c
    const float* row = q0 + (size_t)bc * NSEQ;
    float s = 0.f;
    for (int n = threadIdx.x; n < NSEQ; n += blockDim.x) {
        const float v = row[n];
        s += v * v;
    }
    __shared__ float red[256];
    red[threadIdx.x] = s;
    __syncthreads();
    for (int st = 128; st > 0; st >>= 1) {
        if ((int)threadIdx.x < st) red[threadIdx.x] += red[threadIdx.x + st];
        __syncthreads();
    }
    if (threadIdx.x == 0) qnorm[bc] = sqrtf(red[0]);
}

// ---------------------------------------------------------------------------
// Normalize + convert + pack for the attention WMMA layouts:
//   Qt [b][h][n][ch] : q / ||q||_n        -> B-frags of S (K=ch contiguous)
//   Kt [b][h][n][ch] : k / ||k||_heads    -> A-frags of S (K=ch contiguous)
//   Vf [b][c][n]     : v                  -> B-frags of P*V (K=j contiguous)
// ---------------------------------------------------------------------------
__global__ void pack_f16(const float* __restrict__ q0,
                         const float* __restrict__ k0,
                         const float* __restrict__ v0,
                         const float* __restrict__ qnorm,
                         _Float16* __restrict__ Qt,
                         _Float16* __restrict__ Kt,
                         _Float16* __restrict__ Vf) {
    size_t idx = (size_t)blockIdx.x * blockDim.x + threadIdx.x;
    const size_t total = (size_t)BATCH * DIM * NSEQ;
    if (idx >= total) return;
    const int n = (int)(idx % NSEQ);
    size_t t    = idx / NSEQ;
    const int c = (int)(t % DIM);
    const int b = (int)(t / DIM);
    const int h  = c / CH;
    const int ch = c % CH;
    const size_t src = ((size_t)b * DIM + c) * NSEQ + n;

    const float qn = fmaxf(qnorm[b * DIM + c], 1e-12f);

    float ss = 0.f;
#pragma unroll
    for (int hh = 0; hh < HEADS; ++hh) {
        const float kv = k0[((size_t)b * DIM + hh * CH + ch) * NSEQ + n];
        ss += kv * kv;
    }
    const float kn = fmaxf(sqrtf(ss), 1e-12f);

    const size_t dst = (((size_t)b * HEADS + h) * NSEQ + n) * CH + ch;
    Qt[dst] = (_Float16)(q0[src] / qn);
    Kt[dst] = (_Float16)(k0[src] / kn);
    Vf[src] = (_Float16)(v0[src]);
}

// ---------------------------------------------------------------------------
// Fused flash attention: 4 waves per block cover 64 i-rows of one (b,h).
// Per 32-j step, the Q tile (32x64, 4KB) and V tile (64x32, 4KB) are shared
// by all 4 waves, double-buffered in LDS via async copies.
// Output written directly as f16 [b][n][512] (= B-operand of final GEMM).
// ---------------------------------------------------------------------------
__global__ void flash_attn_wmma(const _Float16* __restrict__ Qt,
                                const _Float16* __restrict__ Kt,
                                const _Float16* __restrict__ Vf,
                                const float* __restrict__ temperature,
                                _Float16* __restrict__ OutT) {
    const int b     = blockIdx.z;
    const int h     = blockIdx.y;
    const int tid   = threadIdx.x;        // 0..127
    const int wave  = tid >> 5;
    const int lane  = tid & 31;
    const int li    = lane & 15;
    const int g     = (lane >> 4) & 1;
    const int ibase = blockIdx.x * 64 + wave * 16;

    const _Float16* Qh = Qt + ((size_t)b * HEADS + h) * NSEQ * CH; // [n][64]
    const _Float16* Kh = Kt + ((size_t)b * HEADS + h) * NSEQ * CH; // [n][64]
    const _Float16* Vh = Vf + ((size_t)b * DIM + h * CH) * NSEQ;   // [ch][n]
    const float temp = temperature[h];

    __shared__ alignas(32) _Float16 qtile[2][32 * 64];   // [j][ch] 4KB x2
    __shared__ alignas(32) _Float16 vtile[2][64 * 32];   // [ch][j] 4KB x2
    __shared__ alignas(32) _Float16 p_lds[4][16 * 32];   // per-wave P tile

    // Stage Q rows jb..jb+31 (contiguous 4KB) and V tile (64 rows x 64B).
    auto stage = [&](int buf, int jb) {
        const _Float16* qsrc = Qh + (size_t)jb * CH;
#pragma unroll
        for (int u = 0; u < 2; ++u) {
            const int chunk = tid * 2 + u;               // 0..255
            async_copy_b128(&qtile[buf][chunk * 8], qsrc + chunk * 8);
        }
#pragma unroll
        for (int u = 0; u < 2; ++u) {
            const int chunk = tid * 2 + u;               // 0..255
            const int row   = chunk >> 2;                // ch
            const int off   = (chunk & 3) * 8;           // j half-offset
            async_copy_b128(&vtile[buf][row * 32 + off],
                            Vh + (size_t)row * NSEQ + jb + off);
        }
    };

    // Fixed A-fragments: K^T rows i, ch = 0..63 (two 32-wide K steps)
    const _Float16* kp = Kh + (size_t)(ibase + li) * CH;
    const v16h aK0 = load_a_frag(kp, g);
    const v16h aK1 = load_a_frag(kp + 32, g);

    v8f o0 = {}, o1 = {}, o2 = {}, o3 = {};
    float m[8], l[8];
#pragma unroll
    for (int r = 0; r < 8; ++r) { m[r] = -1e30f; l[r] = 0.f; }

    stage(0, 0);
    wait_async_barrier();

    for (int jb = 0; jb < NSEQ; jb += 32) {
        const int cur = (jb >> 5) & 1;
        if (jb + 32 < NSEQ) stage(cur ^ 1, jb + 32);

        // ---- S tiles (16 x 32): 4 WMMAs from LDS Q tile ----
        v8f s[2];
#pragma unroll
        for (int jt = 0; jt < 2; ++jt) {
            const _Float16* qrow = &qtile[cur][(jt * 16 + li) * CH];
            const v16h bq0 = load_b_frag(qrow, g);
            const v16h bq1 = load_b_frag(qrow + 32, g);
            v8f z = {};
            z = __builtin_amdgcn_wmma_f32_16x16x32_f16(
                false, aK0, false, bq0, (short)0, z, false, false);
            z = __builtin_amdgcn_wmma_f32_16x16x32_f16(
                false, aK1, false, bq1, (short)0, z, false, false);
            s[jt] = z;
        }

        // ---- online softmax row stats (butterfly within 16-lane half) ----
        float alpha[8];
#pragma unroll
        for (int r = 0; r < 8; ++r) {
            const float v0s = s[0][r] * temp;
            const float v1s = s[1][r] * temp;
            s[0][r] = v0s;
            s[1][r] = v1s;
            float mx = fmaxf(v0s, v1s);
            mx = fmaxf(mx, __shfl_xor(mx, 1));
            mx = fmaxf(mx, __shfl_xor(mx, 2));
            mx = fmaxf(mx, __shfl_xor(mx, 4));
            mx = fmaxf(mx, __shfl_xor(mx, 8));
            const float mnew = fmaxf(m[r], mx);
            alpha[r] = __expf(m[r] - mnew);
            m[r] = mnew;
        }
#pragma unroll
        for (int r = 0; r < 8; ++r) {
            const float p0 = __expf(s[0][r] - m[r]);
            const float p1 = __expf(s[1][r] - m[r]);
            s[0][r] = p0;
            s[1][r] = p1;
            float rs = p0 + p1;
            rs += __shfl_xor(rs, 1);
            rs += __shfl_xor(rs, 2);
            rs += __shfl_xor(rs, 4);
            rs += __shfl_xor(rs, 8);
            l[r] = l[r] * alpha[r] + rs;
            o0[r] *= alpha[r];
            o1[r] *= alpha[r];
            o2[r] *= alpha[r];
            o3[r] *= alpha[r];
        }

        // ---- stage P through per-wave LDS (wave-internal DS ordering) ----
#pragma unroll
        for (int r = 0; r < 8; ++r) {
            const int row = r + 8 * g;
            p_lds[wave][row * 32 + li]      = (_Float16)s[0][r];
            p_lds[wave][row * 32 + 16 + li] = (_Float16)s[1][r];
        }
        const v16h aP = load_a_frag(&p_lds[wave][li * 32], g);

        // ---- O += P * V^T : 4 WMMAs from LDS V tile ----
#pragma unroll
        for (int ct = 0; ct < 4; ++ct) {
            const v16h bv = load_b_frag(&vtile[cur][(ct * 16 + li) * 32], g);
            v8f* op = (ct == 0) ? &o0 : (ct == 1) ? &o1 : (ct == 2) ? &o2 : &o3;
            *op = __builtin_amdgcn_wmma_f32_16x16x32_f16(
                false, aP, false, bv, (short)0, *op, false, false);
        }
        wait_async_barrier();                 // next Q/V tiles landed
    }

    // ---- finalize: /l, write f16 transposed [b][i][h*64+c] ----
#pragma unroll
    for (int r = 0; r < 8; ++r) {
        const int i = ibase + r + 8 * g;
        const float inv = 1.f / fmaxf(l[r], 1e-30f);
        _Float16* orow = OutT + ((size_t)b * NSEQ + i) * DIM + h * CH;
        orow[0 * 16 + li] = (_Float16)(o0[r] * inv);
        orow[1 * 16 + li] = (_Float16)(o1[r] * inv);
        orow[2 * 16 + li] = (_Float16)(o2[r] * inv);
        orow[3 * 16 + li] = (_Float16)(o3[r] * inv);
    }
}

// ---------------------------------------------------------------------------
// Host-side orchestration
// ---------------------------------------------------------------------------
extern "C" void kernel_launch(void* const* d_in, const int* in_sizes, int n_in,
                              void* d_out, int out_size, void* d_ws, size_t ws_size,
                              hipStream_t stream) {
    const float* x     = (const float*)d_in[0]; // [B, 512, 2048]
    const float* w_qkv = (const float*)d_in[1]; // [1536, 512, 1]
    const float* b_qkv = (const float*)d_in[2]; // [1536]
    const float* w_dw  = (const float*)d_in[3]; // [1536, 1, 3]
    const float* b_dw  = (const float*)d_in[4]; // [1536]
    const float* w_po  = (const float*)d_in[5]; // [512, 512, 1]
    const float* b_po  = (const float*)d_in[6]; // [512]
    const float* temp  = (const float*)d_in[7]; // [8, 1, 1]
    float* out = (float*)d_out;                 // [B, 512, 2048]

    char* ws = (char*)d_ws;
    const size_t sz_qkv  = (size_t)BATCH * C3  * NSEQ * sizeof(float);
    const size_t sz_f32  = (size_t)BATCH * DIM * NSEQ * sizeof(float);
    const size_t sz_f16  = (size_t)BATCH * DIM * NSEQ * sizeof(_Float16);
    const size_t sz_norm = (size_t)BATCH * DIM * sizeof(float);
    const size_t sz_wq16 = (size_t)C3 * DIM * sizeof(_Float16);
    const size_t sz_wp16 = (size_t)DIM * DIM * sizeof(_Float16);

    float*    qkv   = (float*)ws;     ws += sz_qkv;
    float*    q0    = (float*)ws;     ws += sz_f32;
    float*    k0    = (float*)ws;     ws += sz_f32;
    float*    v0    = (float*)ws;     ws += sz_f32;
    float*    qnorm = (float*)ws;     ws += sz_norm;
    _Float16* Qt    = (_Float16*)ws;  ws += sz_f16;
    _Float16* Kt    = (_Float16*)ws;  ws += sz_f16;
    _Float16* Vf    = (_Float16*)ws;  ws += sz_f16;
    _Float16* Xt    = (_Float16*)ws;  ws += sz_f16;
    _Float16* OutT  = (_Float16*)ws;  ws += sz_f16;
    _Float16* wq16  = (_Float16*)ws;  ws += sz_wq16;
    _Float16* wp16  = (_Float16*)ws;  ws += sz_wp16;

    // 0) weight + activation conversion / transpose
    cvt_f16_flat<<<(C3 * DIM + 255) / 256, 256, 0, stream>>>(w_qkv, wq16, C3 * DIM);
    cvt_f16_flat<<<(DIM * DIM + 255) / 256, 256, 0, stream>>>(w_po, wp16, DIM * DIM);
    {
        const size_t total = (size_t)BATCH * NSEQ * DIM;
        transpose_cvt_x<<<(unsigned)((total + 255) / 256), 256, 0, stream>>>(x, Xt);
    }

    // 1) qkv = W_qkv @ x + b_qkv    [1536 x 512] @ [512 x 2048] per batch
    gemm_pw_wmma<<<dim3(C3 / 64, NSEQ / 128, BATCH), 128, 0, stream>>>(
        wq16, b_qkv, Xt, qkv, C3, DIM, NSEQ);

    // 2) depthwise conv + split
    {
        const size_t total = (size_t)BATCH * C3 * NSEQ;
        dwconv_split<<<(unsigned)((total + 255) / 256), 256, 0, stream>>>(
            qkv, w_dw, b_dw, q0, k0, v0);
    }

    // 3) q row norms
    qnorm_kernel<<<BATCH * DIM, 256, 0, stream>>>(q0, qnorm);

    // 4) normalize + pack f16 WMMA layouts
    {
        const size_t total = (size_t)BATCH * DIM * NSEQ;
        pack_f16<<<(unsigned)((total + 255) / 256), 256, 0, stream>>>(
            q0, k0, v0, qnorm, Qt, Kt, Vf);
    }

    // 5) fused flash attention -> f16 [b][n][512]
    flash_attn_wmma<<<dim3(NSEQ / 64, HEADS, BATCH), 128, 0, stream>>>(
        Qt, Kt, Vf, temp, OutT);

    // 6) out = W_po @ attn_out + b_po   [512 x 512] @ [512 x 2048] per batch
    gemm_pw_wmma<<<dim3(DIM / 64, NSEQ / 128, BATCH), 128, 0, stream>>>(
        wp16, b_po, OutT, out, DIM, DIM, NSEQ);
}
